// GraphSAGE_66709432041918
// MI455X (gfx1250) — matrix operand order repaired and checked by reading, not compiled, not run on hardware
//
#include <hip/hip_runtime.h>
#include <hip/hip_bf16.h>

// ---------------------------------------------------------------------------
// GraphSAGE (3-layer) for MI455X / gfx1250, wave32.
// Edge aggregation = bandwidth-bound (L2-resident, f32 atomics).
// Dense layers = v_wmma_f32_16x16x32_bf16 (f32 accumulate).
// ---------------------------------------------------------------------------

#define N_NODES 100000
#define N_EDGES 1600000
#define DIM     128

typedef __bf16 bf16_t;
typedef __attribute__((ext_vector_type(16))) __bf16 v16bf;
typedef __attribute__((ext_vector_type(8)))  __bf16 v8bf;
typedef __attribute__((ext_vector_type(8)))  float  v8f;

// ---------------- utility kernels ----------------

__global__ void zero_f32_kernel(float* __restrict__ p, int n) {
    int i = blockIdx.x * blockDim.x + threadIdx.x;
    if (i < n) p[i] = 0.0f;
}

// count in-degree (accumulate 1.0 per edge into deg buffer)
__global__ void count_deg_kernel(const long long* __restrict__ dst,
                                 float* __restrict__ deg) {
    int e = blockIdx.x * blockDim.x + threadIdx.x;
    if (e < N_EDGES) atomicAdd(&deg[(int)dst[e]], 1.0f);
}

// in-place: deg -> 1/max(deg,1)
__global__ void deg_inv_kernel(float* __restrict__ deg) {
    int i = blockIdx.x * blockDim.x + threadIdx.x;
    if (i < N_NODES) deg[i] = 1.0f / fmaxf(deg[i], 1.0f);
}

// ---------------- edge aggregation (dominant cost) ----------------
// One wave per edge; lane handles 4 consecutive features (float4 gather,
// 4x global_atomic_add_f32 scatter). Features + accumulator are L2-resident.
__global__ __launch_bounds__(256) void aggregate_kernel(
    const float* __restrict__ feat,
    const long long* __restrict__ src,
    const long long* __restrict__ dst,
    float* __restrict__ agg)
{
    int gid  = blockIdx.x * blockDim.x + threadIdx.x;
    int edge = gid >> 5;
    int lane = threadIdx.x & 31;
    if (edge >= N_EDGES) return;
    int s = (int)src[edge];
    int d = (int)dst[edge];
    const float4 v = *(const float4*)(feat + (size_t)s * DIM + lane * 4);
    float* a = agg + (size_t)d * DIM + lane * 4;
    atomicAdd(a + 0, v.x);
    atomicAdd(a + 1, v.y);
    atomicAdd(a + 2, v.z);
    atomicAdd(a + 3, v.w);
}

// scale agg by deg_inv and convert both agg and the layer-input features
// to bf16 for the WMMA GEMM
__global__ __launch_bounds__(256) void scale_convert_kernel(
    const float* __restrict__ agg,
    const float* __restrict__ feat,
    const float* __restrict__ dinv,
    bf16_t* __restrict__ aggb,
    bf16_t* __restrict__ featb)
{
    long long i = (long long)blockIdx.x * blockDim.x + threadIdx.x;
    if (i >= (long long)N_NODES * DIM) return;
    int node = (int)(i >> 7);
    float s = dinv[node];
    aggb[i]  = (bf16_t)(agg[i] * s);
    featb[i] = (bf16_t)feat[i];
}

// W (128x128 f32, row-major [k][n]) -> WT (bf16, [n][k]) so the WMMA B-matrix
// per-lane K-run is a contiguous 32B load.
__global__ void wtranspose_kernel(const float* __restrict__ W,
                                  bf16_t* __restrict__ WT) {
    int i = blockIdx.x * blockDim.x + threadIdx.x;   // 16384
    if (i >= DIM * DIM) return;
    int n = i >> 7, k = i & 127;
    WT[i] = (bf16_t)W[k * DIM + n];
}

// ---------------- WMMA GEMM: out = relu(aggb@Wl + xb@Wr + b) ----------------
// Block = 8 waves; wave w computes a 16x16 tile at (m0 = blockIdx.x*16,
// n0 = w*16). K loop: 4 steps of 32, two WMMAs per step (agg path + x path).
__global__ __launch_bounds__(256) void sage_gemm_kernel(
    const bf16_t* __restrict__ aggb,
    const bf16_t* __restrict__ xb,
    const bf16_t* __restrict__ WlT,   // [n][k] bf16
    const bf16_t* __restrict__ WrT,   // [n][k] bf16
    const float*  __restrict__ bias,  // [128]
    float* __restrict__ out)          // [N][128] f32
{
    const int lane = threadIdx.x;          // 0..31
    const int m0   = blockIdx.x * 16;      // 6250 tiles, exact
    const int n0   = threadIdx.y * 16;     // 8 waves -> 128 cols
    const bool hi  = lane >= 16;
    const int  l   = lane & 15;

    // A-matrix (16x32 bf16) per-lane chunks: lanes 0-15 hold K 0..15,
    // lanes 16-31 hold K 16..31 of row M=l.
    const int aoff = hi ? 16 : 0;
    const bf16_t* arow = aggb + (size_t)(m0 + l) * DIM;
    const bf16_t* xrow = xb   + (size_t)(m0 + l) * DIM;
    // B-matrix (32x16 bf16): lane = column n0+l; lanes 0-15 take K 0..15,
    // lanes 16-31 take K 16..31 (contiguous in [n][k] layout).
    const bf16_t* blrow = WlT + (size_t)(n0 + l) * DIM + (hi ? 16 : 0);
    const bf16_t* brrow = WrT + (size_t)(n0 + l) * DIM + (hi ? 16 : 0);

    v8f acc = {};
#pragma unroll
    for (int k0 = 0; k0 < DIM; k0 += 32) {
        // agg path
        v8bf a0 = *(const v8bf*)(arow + k0 + aoff);
        v8bf a1 = *(const v8bf*)(arow + k0 + 8 + aoff);
        v16bf a;
#pragma unroll
        for (int t = 0; t < 8; ++t) { a[t] = a0[t]; a[t + 8] = a1[t]; }
        v16bf b = *(const v16bf*)(blrow + k0);
        acc = __builtin_amdgcn_wmma_f32_16x16x32_bf16(
            false, a, false, b, (short)0, acc, false, false);

        // x (self) path
        v8bf c0 = *(const v8bf*)(xrow + k0 + aoff);
        v8bf c1 = *(const v8bf*)(xrow + k0 + 8 + aoff);
        v16bf a2;
#pragma unroll
        for (int t = 0; t < 8; ++t) { a2[t] = c0[t]; a2[t + 8] = c1[t]; }
        v16bf b2 = *(const v16bf*)(brrow + k0);
        acc = __builtin_amdgcn_wmma_f32_16x16x32_bf16(
            false, a2, false, b2, (short)0, acc, false, false);
    }

    // C/D layout: VGPR r -> M = r (+8 for lanes 16-31), N = lane%16
    const int nn = n0 + l;
    const float bv = bias[nn];
#pragma unroll
    for (int r = 0; r < 8; ++r) {
        int m = m0 + r + (hi ? 8 : 0);
        float v = acc[r] + bv;
        v = fmaxf(v, 0.0f);
        out[(size_t)m * DIM + nn] = v;
    }
}

// ---------------- layer 3 (128 -> 2) + relu + log_softmax ----------------
// One wave per node: lane-parallel dot products, shuffle reduce.
__global__ __launch_bounds__(256) void sage_out_kernel(
    const float* __restrict__ agg,
    const float* __restrict__ h,
    const float* __restrict__ dinv,
    const float* __restrict__ W3l,   // [128][2] f32
    const float* __restrict__ W3r,   // [128][2] f32
    const float* __restrict__ b3,    // [2]
    float* __restrict__ out)         // [N][2]
{
    int gid  = blockIdx.x * blockDim.x + threadIdx.x;
    int node = gid >> 5;
    int lane = threadIdx.x & 31;
    if (node >= N_NODES) return;

    float s = dinv[node];
    const float* arow = agg + (size_t)node * DIM;
    const float* hrow = h   + (size_t)node * DIM;
    float s0 = 0.0f, s1 = 0.0f;
#pragma unroll
    for (int kk = 0; kk < DIM; kk += 32) {
        int k = kk + lane;
        float av = arow[k] * s;
        float hv = hrow[k];
        s0 += av * W3l[k * 2 + 0] + hv * W3r[k * 2 + 0];
        s1 += av * W3l[k * 2 + 1] + hv * W3r[k * 2 + 1];
    }
#pragma unroll
    for (int off = 16; off > 0; off >>= 1) {
        s0 += __shfl_down(s0, off, 32);
        s1 += __shfl_down(s1, off, 32);
    }
    if (lane == 0) {
        float z0 = fmaxf(s0 + b3[0], 0.0f);
        float z1 = fmaxf(s1 + b3[1], 0.0f);
        float m  = fmaxf(z0, z1);
        float lse = m + logf(__expf(z0 - m) + __expf(z1 - m));
        out[(size_t)node * 2 + 0] = z0 - lse;
        out[(size_t)node * 2 + 1] = z1 - lse;
    }
}

// ---------------- host orchestration ----------------

extern "C" void kernel_launch(void* const* d_in, const int* in_sizes, int n_in,
                              void* d_out, int out_size, void* d_ws, size_t ws_size,
                              hipStream_t stream) {
    (void)in_sizes; (void)n_in; (void)out_size; (void)ws_size;

    const float*     x   = (const float*)d_in[0];
    const long long* ei  = (const long long*)d_in[1];   // int64 [2][E]
    const long long* src = ei;
    const long long* dst = ei + N_EDGES;
    const float* W1l = (const float*)d_in[2];
    const float* W1r = (const float*)d_in[3];
    const float* b1  = (const float*)d_in[4];
    const float* W2l = (const float*)d_in[5];
    const float* W2r = (const float*)d_in[6];
    const float* b2  = (const float*)d_in[7];
    const float* W3l = (const float*)d_in[8];
    const float* W3r = (const float*)d_in[9];
    const float* b3  = (const float*)d_in[10];
    float* out = (float*)d_out;

    // workspace carve-out (256B aligned)
    size_t off = 0;
    char* base = (char*)d_ws;
    auto carve = [&](size_t bytes) -> void* {
        void* p = base + off;
        off += (bytes + 255) & ~(size_t)255;
        return p;
    };
    const size_t NF = (size_t)N_NODES * DIM;
    float*  dinv = (float*)carve(N_NODES * sizeof(float));
    float*  agg  = (float*)carve(NF * sizeof(float));
    float*  h1   = (float*)carve(NF * sizeof(float));
    float*  h2   = (float*)carve(NF * sizeof(float));
    bf16_t* aggb = (bf16_t*)carve(NF * sizeof(bf16_t));
    bf16_t* xb   = (bf16_t*)carve(NF * sizeof(bf16_t));
    bf16_t* WlT  = (bf16_t*)carve(DIM * DIM * sizeof(bf16_t));
    bf16_t* WrT  = (bf16_t*)carve(DIM * DIM * sizeof(bf16_t));

    const int gNode  = (N_NODES + 255) / 256;
    const int gFeat  = (int)((NF + 255) / 256);
    const int gEdgeT = (N_EDGES + 255) / 256;            // thread-per-edge
    const int gEdgeW = (int)(((size_t)N_EDGES * 32 + 255) / 256); // wave-per-edge
    const int gOutW  = (int)(((size_t)N_NODES * 32 + 255) / 256);
    const dim3 gemmBlk(32, 8);

    // degrees
    zero_f32_kernel<<<gNode, 256, 0, stream>>>(dinv, N_NODES);
    count_deg_kernel<<<gEdgeT, 256, 0, stream>>>(dst, dinv);
    deg_inv_kernel<<<gNode, 256, 0, stream>>>(dinv);

    // ---- layer 1 ----
    zero_f32_kernel<<<gFeat, 256, 0, stream>>>(agg, (int)NF);
    aggregate_kernel<<<gEdgeW, 256, 0, stream>>>(x, src, dst, agg);
    scale_convert_kernel<<<gFeat, 256, 0, stream>>>(agg, x, dinv, aggb, xb);
    wtranspose_kernel<<<64, 256, 0, stream>>>(W1l, WlT);
    wtranspose_kernel<<<64, 256, 0, stream>>>(W1r, WrT);
    sage_gemm_kernel<<<N_NODES / 16, gemmBlk, 0, stream>>>(aggb, xb, WlT, WrT, b1, h1);

    // ---- layer 2 ----
    zero_f32_kernel<<<gFeat, 256, 0, stream>>>(agg, (int)NF);
    aggregate_kernel<<<gEdgeW, 256, 0, stream>>>(h1, src, dst, agg);
    scale_convert_kernel<<<gFeat, 256, 0, stream>>>(agg, h1, dinv, aggb, xb);
    wtranspose_kernel<<<64, 256, 0, stream>>>(W2l, WlT);
    wtranspose_kernel<<<64, 256, 0, stream>>>(W2r, WrT);
    sage_gemm_kernel<<<N_NODES / 16, gemmBlk, 0, stream>>>(aggb, xb, WlT, WrT, b2, h2);

    // ---- layer 3 + log_softmax ----
    zero_f32_kernel<<<gFeat, 256, 0, stream>>>(agg, (int)NF);
    aggregate_kernel<<<gEdgeW, 256, 0, stream>>>(h2, src, dst, agg);
    sage_out_kernel<<<gOutW, 256, 0, stream>>>(agg, h2, dinv, W3l, W3r, b3, out);
}
